// IHBNet_54460185313877
// MI455X (gfx1250) — compile-verified, matrix operand order checked by reading, not run on hardware
//
#include <hip/hip_runtime.h>
#include <cstddef>
#include <cstdint>

// ---------------- problem constants ----------------
#define EMBED   768
#define TOKENS  197
#define BATCH   8
#define NPATCH  196
#define MLPDIM  3072
#define NHEADS  12
#define HDIM    64
#define MREAL_T (BATCH*TOKENS)   // 1576 rows (transformer)
#define MREAL_P (BATCH*NPATCH)   // 1568 rows (patch / graph)
#define MPAD    1600             // padded M, multiple of 64

// ---------------- WMMA types ----------------
typedef __attribute__((ext_vector_type(16))) __bf16 v16bf;
typedef __attribute__((ext_vector_type(8)))  float  v8f;

union AFrag { v16bf v; uint4 q[2]; __bf16 e[16]; };

__device__ __forceinline__ float gelu_exact(float x) {
    return 0.5f * x * (1.0f + erff(x * 0.70710678118654752440f));
}

__device__ __forceinline__ float block_reduce_sum(float v, float* red) {
    int t = threadIdx.x;
    red[t] = v; __syncthreads();
    for (int s = blockDim.x >> 1; s > 0; s >>= 1) {
        if (t < s) red[t] += red[t + s];
        __syncthreads();
    }
    float r = red[0]; __syncthreads();
    return r;
}

// convert 8 consecutive fp32 weights to 8 bf16 fragment elements
__device__ __forceinline__ void cvt8(const float* __restrict__ p, __bf16* __restrict__ d) {
    float4 a = *(const float4*)(p);
    float4 b = *(const float4*)(p + 4);
    d[0] = (__bf16)a.x; d[1] = (__bf16)a.y; d[2] = (__bf16)a.z; d[3] = (__bf16)a.w;
    d[4] = (__bf16)b.x; d[5] = (__bf16)b.y; d[6] = (__bf16)b.z; d[7] = (__bf16)b.w;
}

// CDNA5 async copy: 16 bytes global -> LDS per lane, tracked by ASYNCcnt
__device__ __forceinline__ void async_ld16(unsigned ldsOff, const __bf16* g) {
    asm volatile("global_load_async_to_lds_b128 %0, %1, off"
                 :: "v"(ldsOff), "v"(g)
                 : "memory");
}
__device__ __forceinline__ void wait_asynccnt0() {
    asm volatile("s_wait_asynccnt 0" ::: "memory");
}

// ---------------- WMMA GEMM ----------------
// C[MPAD,N] = act( A[MPAD,K](bf16) @ W[N,K](f32, converted to bf16 in-register)^T + bias )
// grid = (N/64, MPAD/64), block = 128 threads = 4 wave32; each wave owns a 32x32 C tile.
// A block-tile (64 rows x 32 k, bf16) is staged in LDS via async global->LDS copies.
template<int ACT, int OUTBF>
__global__ __launch_bounds__(128)
void k_gemm(const __bf16* __restrict__ A, const float* __restrict__ W,
            const float* __restrict__ bias, void* __restrict__ Cv, int N, int K) {
    __shared__ alignas(16) __bf16 sA[64 * 32];   // row stride 32 elems = 64 B

    const int tid  = threadIdx.x;
    const int lane = tid & 31;
    const int wave = tid >> 5;
    const int hl   = lane >> 4;     // half-wave selects K sub-groups (ISA 16-bit A layout)
    const int lm   = lane & 15;     // M row (A) / N column (B) within tile

    const int mBlk = blockIdx.y * 64;
    const int m0   = (wave >> 1) * 32;                 // local A-row base for this wave
    const int n0   = blockIdx.x * 64 + (wave & 1) * 32;

    const float* wrow0 = W + (size_t)(n0 + lm) * K;
    const float* wrow1 = W + (size_t)(n0 + 16 + lm) * K;

    // async-copy mapping: 128 threads x 2 chunks x 16 B = 4 KB tile
    const int id0 = tid * 2, id1 = tid * 2 + 1;
    const int r0 = id0 >> 2, c0 = id0 & 3;
    const int r1 = id1 >> 2, c1 = id1 & 3;
    const unsigned sBase = (unsigned)(uintptr_t)(&sA[0]);   // LDS offset = low 32 bits
    const unsigned ld0 = sBase + (unsigned)(r0 * 64 + c0 * 16);
    const unsigned ld1 = sBase + (unsigned)(r1 * 64 + c1 * 16);
    const __bf16* g0 = A + (size_t)(mBlk + r0) * K + c0 * 8;
    const __bf16* g1 = A + (size_t)(mBlk + r1) * K + c1 * 8;

    const __bf16* sRow0 = &sA[(m0 + lm) * 32 + hl * 8];
    const __bf16* sRow1 = &sA[(m0 + 16 + lm) * 32 + hl * 8];

    v8f acc00 = {}, acc01 = {}, acc10 = {}, acc11 = {};

    for (int k = 0; k < K; k += 32) {
        // stage A tile in LDS (async DMA path, ASYNCcnt)
        async_ld16(ld0, g0 + k);
        async_ld16(ld1, g1 + k);
        wait_asynccnt0();
        __syncthreads();

        const int kb = k + 8 * hl;
        AFrag a0, a1, b0, b1;
        a0.q[0] = *(const uint4*)(sRow0);        // ds_load_b128
        a0.q[1] = *(const uint4*)(sRow0 + 16);
        a1.q[0] = *(const uint4*)(sRow1);
        a1.q[1] = *(const uint4*)(sRow1 + 16);
        cvt8(wrow0 + kb,      b0.e);
        cvt8(wrow0 + kb + 16, b0.e + 8);
        cvt8(wrow1 + kb,      b1.e);
        cvt8(wrow1 + kb + 16, b1.e + 8);
        __builtin_prefetch(wrow0 + kb + 32, 0, 1);   // global_prefetch on weight stream

        acc00 = __builtin_amdgcn_wmma_f32_16x16x32_bf16(false, a0.v, false, b0.v, (short)0, acc00, false, false);
        acc01 = __builtin_amdgcn_wmma_f32_16x16x32_bf16(false, a0.v, false, b1.v, (short)0, acc01, false, false);
        acc10 = __builtin_amdgcn_wmma_f32_16x16x32_bf16(false, a1.v, false, b0.v, (short)0, acc10, false, false);
        acc11 = __builtin_amdgcn_wmma_f32_16x16x32_bf16(false, a1.v, false, b1.v, (short)0, acc11, false, false);

        __syncthreads();   // tile consumed before next overwrite
    }

    const float bn0 = bias ? bias[n0 + lm]      : 0.0f;
    const float bn1 = bias ? bias[n0 + 16 + lm] : 0.0f;

#define GEMM_STORE(mr, nc, val) do {                                          \
        float _v = (val);                                                     \
        if (ACT == 1) _v = gelu_exact(_v);                                    \
        if (OUTBF) ((__bf16*)Cv)[(size_t)(mr) * N + (nc)] = (__bf16)_v;       \
        else       ((float*) Cv)[(size_t)(mr) * N + (nc)] = _v;               \
    } while (0)

#pragma unroll
    for (int r = 0; r < 8; ++r) {
        const int mr0 = mBlk + m0 + r + 8 * hl;      // C/D layout: vgpr r -> M=r (+8 hi half)
        const int mr1 = mBlk + m0 + 16 + r + 8 * hl;
        GEMM_STORE(mr0, n0 + lm,      acc00[r] + bn0);
        GEMM_STORE(mr0, n0 + 16 + lm, acc01[r] + bn1);
        GEMM_STORE(mr1, n0 + lm,      acc10[r] + bn0);
        GEMM_STORE(mr1, n0 + 16 + lm, acc11[r] + bn1);
    }
#undef GEMM_STORE
}

// ---------------- im2col: 16x16 stride-16 patches ----------------
__global__ void k_im2col_patch(const float* __restrict__ x, __bf16* __restrict__ A) {
    int idx = blockIdx.x * 256 + threadIdx.x;
    if (idx >= MREAL_P * EMBED) return;
    int row = idx / EMBED, col = idx - row * EMBED;
    int b = row / NPATCH, p = row - b * NPATCH;
    int py = p / 14, px = p - py * 14;
    int c = col >> 8, r = col & 255;
    int dy = r >> 4, dx = r & 15;
    float v = x[(((size_t)b * 3 + c) * 224 + (py * 16 + dy)) * 224 + (px * 16 + dx)];
    A[(size_t)row * EMBED + col] = (__bf16)v;
}

// ---------------- im2col: 3x3 SAME conv over y[:,1:,:] ----------------
__global__ void k_im2col_sp(const float* __restrict__ y, __bf16* __restrict__ A) {
    int idx = blockIdx.x * 256 + threadIdx.x;
    if (idx >= MREAL_P * EMBED * 9) return;
    int row = idx / (EMBED * 9), col = idx - row * (EMBED * 9);
    int b = row / NPATCH, p = row - b * NPATCH;
    int py = p / 14, px = p - py * 14;
    int e = col / 9, kk = col - e * 9;
    int ky = kk / 3 - 1, kx = kk % 3 - 1;
    int sy = py + ky, sx = px + kx;
    float v = 0.0f;
    if (sy >= 0 && sy < 14 && sx >= 0 && sx < 14)
        v = y[((size_t)b * TOKENS + 1 + sy * 14 + sx) * EMBED + e];
    A[(size_t)row * (EMBED * 9) + col] = (__bf16)v;
}

// ---------------- h = concat(cls, patches) + pos ----------------
__global__ void k_build_h(const float* __restrict__ P, const float* __restrict__ cls,
                          const float* __restrict__ pos, float* __restrict__ H) {
    int idx = blockIdx.x * 256 + threadIdx.x;
    if (idx >= MREAL_T * EMBED) return;
    int e = idx % EMBED, bt = idx / EMBED;
    int t = bt % TOKENS, b = bt / TOKENS;
    float base = (t == 0) ? cls[e] : P[((size_t)b * NPATCH + (t - 1)) * EMBED + e];
    H[idx] = base + pos[t * EMBED + e];
}

__global__ void k_resadd(float* __restrict__ H, const float* __restrict__ S, int n) {
    int i = blockIdx.x * 256 + threadIdx.x;
    if (i < n) H[i] += S[i];
}

// ---------------- LayerNorm -> bf16 ----------------
__global__ void k_layernorm(const float* __restrict__ X, const float* __restrict__ g,
                            const float* __restrict__ bta, __bf16* __restrict__ Y) {
    __shared__ float red[256];
    int row = blockIdx.x;
    const float* x = X + (size_t)row * EMBED;
    float s = 0.0f;
    for (int i = threadIdx.x; i < EMBED; i += 256) s += x[i];
    float mean = block_reduce_sum(s, red) * (1.0f / EMBED);
    float v = 0.0f;
    for (int i = threadIdx.x; i < EMBED; i += 256) { float d = x[i] - mean; v += d * d; }
    float rs = rsqrtf(block_reduce_sum(v, red) * (1.0f / EMBED) + 1e-6f);
    __bf16* y = Y + (size_t)row * EMBED;
    for (int i = threadIdx.x; i < EMBED; i += 256)
        y[i] = (__bf16)((x[i] - mean) * rs * g[i] + bta[i]);
}

// ---------------- final: LN, +pos, LN again; emit y and cls_out ----------------
__global__ void k_final_norm(const float* __restrict__ H, const float* __restrict__ g,
                             const float* __restrict__ bta, const float* __restrict__ pos,
                             float* __restrict__ Y, float* __restrict__ clsOut) {
    __shared__ float red[256];
    __shared__ float buf[EMBED];
    int row = blockIdx.x;
    int t = row % TOKENS, b = row / TOKENS;
    const float* x = H + (size_t)row * EMBED;
    float s = 0.0f;
    for (int i = threadIdx.x; i < EMBED; i += 256) s += x[i];
    float mean = block_reduce_sum(s, red) * (1.0f / EMBED);
    float v = 0.0f;
    for (int i = threadIdx.x; i < EMBED; i += 256) { float d = x[i] - mean; v += d * d; }
    float rs = rsqrtf(block_reduce_sum(v, red) * (1.0f / EMBED) + 1e-6f);
    for (int i = threadIdx.x; i < EMBED; i += 256)
        buf[i] = (x[i] - mean) * rs * g[i] + bta[i] + pos[t * EMBED + i];
    __syncthreads();
    s = 0.0f;
    for (int i = threadIdx.x; i < EMBED; i += 256) s += buf[i];
    mean = block_reduce_sum(s, red) * (1.0f / EMBED);
    v = 0.0f;
    for (int i = threadIdx.x; i < EMBED; i += 256) { float d = buf[i] - mean; v += d * d; }
    rs = rsqrtf(block_reduce_sum(v, red) * (1.0f / EMBED) + 1e-6f);
    for (int i = threadIdx.x; i < EMBED; i += 256) {
        float o = (buf[i] - mean) * rs * g[i] + bta[i];
        Y[(size_t)row * EMBED + i] = o;
        if (t == 0) clsOut[(size_t)b * EMBED + i] = o;
    }
}

// ---------------- attention: softmax(q k^T / 8) v, per (b,h,q) ----------------
__global__ void k_attention(const float* __restrict__ qkv, __bf16* __restrict__ O) {
    __shared__ float sc[256];
    __shared__ float red[256];
    __shared__ float qv[HDIM];
    int blk = blockIdx.x;                  // (b*NHEADS + h)*TOKENS + t
    int t = blk % TOKENS, bh = blk / TOKENS;
    int h = bh % NHEADS, b = bh / NHEADS;
    const size_t ld = 3 * EMBED;
    const float* qrow = qkv + ((size_t)(b * TOKENS + t)) * ld + h * HDIM;
    if (threadIdx.x < HDIM) qv[threadIdx.x] = qrow[threadIdx.x];
    __syncthreads();
    int j = threadIdx.x;
    float e = -1e30f;
    if (j < TOKENS) {
        const float* krow = qkv + ((size_t)(b * TOKENS + j)) * ld + EMBED + h * HDIM;
        float d = 0.0f;
        for (int i = 0; i < HDIM; ++i) d += qv[i] * krow[i];
        e = d * 0.125f;
    }
    sc[j] = e;
    red[j] = e; __syncthreads();
    for (int s = 128; s > 0; s >>= 1) { if (j < s) red[j] = fmaxf(red[j], red[j + s]); __syncthreads(); }
    float m = red[0]; __syncthreads();
    float ex = (j < TOKENS) ? __expf(sc[j] - m) : 0.0f;
    sc[j] = ex;
    red[j] = ex; __syncthreads();
    for (int s = 128; s > 0; s >>= 1) { if (j < s) red[j] += red[j + s]; __syncthreads(); }
    float inv = 1.0f / red[0];
    __syncthreads();
    if (j < HDIM) {
        float acc = 0.0f;
        const float* vbase = qkv + ((size_t)(b * TOKENS)) * ld + 2 * EMBED + h * HDIM + j;
        for (int k2 = 0; k2 < TOKENS; ++k2) acc += sc[k2] * vbase[(size_t)k2 * ld];
        O[((size_t)(b * TOKENS + t)) * EMBED + h * HDIM + j] = (__bf16)(acc * inv);
    }
}

// ---------------- spatial conv output + gaussian PE -> bf16 nodes ----------------
__global__ void k_pe_add(const float* __restrict__ S, const float* __restrict__ gauss,
                         __bf16* __restrict__ Nd) {
    int idx = blockIdx.x * 256 + threadIdx.x;
    if (idx >= MREAL_P * EMBED) return;
    int row = idx / EMBED, e = idx - row * EMBED;
    int p = row % NPATCH;
    int yy = p / 14, xx = p - yy * 14;
    float cx = ((xx + 0.5f) / 14.0f) * 2.0f - 1.0f;
    float cy = ((yy + 0.5f) / 14.0f) * 2.0f - 1.0f;
    int j = (e < 384) ? e : e - 384;
    float f = 6.28318530717958648f * (cx * gauss[j] + cy * gauss[384 + j]);
    float pe = (e < 384) ? sinf(f) : cosf(f);
    Nd[idx] = (__bf16)(S[idx] + pe);
}

// ---------------- GAT attention scores per node/head ----------------
__global__ void k_gat_scores(const float* __restrict__ H, const float* __restrict__ asrc,
                             const float* __restrict__ adst, float* __restrict__ Ss,
                             float* __restrict__ Sd) {
    __shared__ float red[256];
    int row = blockIdx.x;                         // b*196 + n
    const float* hr = H + (size_t)row * (4 * EMBED);
    for (int gh = 0; gh < 4; ++gh) {
        float s1 = 0.0f, s2 = 0.0f;
        for (int i = threadIdx.x; i < EMBED; i += 256) {
            float hv = hr[gh * EMBED + i];
            s1 += hv * asrc[gh * EMBED + i];
            s2 += hv * adst[gh * EMBED + i];
        }
        float r1 = block_reduce_sum(s1, red);
        float r2 = block_reduce_sum(s2, red);
        if (threadIdx.x == 0) { Ss[row * 4 + gh] = r1; Sd[row * 4 + gh] = r2; }
        __syncthreads();
    }
}

// ---------------- GAT aggregation over 3x3 neighborhood ----------------
__global__ void k_gat_agg(const float* __restrict__ H, const float* __restrict__ Ss,
                          const float* __restrict__ Sd, const float* __restrict__ gbias,
                          __bf16* __restrict__ OG) {
    __shared__ float alpha[9];
    __shared__ int   srcrow[9];
    __shared__ int   nvalid;
    int blk = blockIdx.x;                         // (b*196 + n)*4 + gh
    int gh = blk & 3, rn = blk >> 2;
    int n = rn % NPATCH, b = rn / NPATCH;
    int yy = n / 14, xx = n % 14;
    if (threadIdx.x == 0) {
        float ev[9]; int sr[9]; int cnt = 0; float mx = -1e30f;
        float sdv = Sd[rn * 4 + gh];
        for (int oy = -1; oy <= 1; ++oy)
            for (int ox = -1; ox <= 1; ++ox) {
                int sy = yy + oy, sx = xx + ox;
                if (sy < 0 || sy > 13 || sx < 0 || sx > 13) continue;
                int s = b * NPATCH + sy * 14 + sx;
                float e = Ss[s * 4 + gh] + sdv;
                e = (e >= 0.0f) ? e : 0.2f * e;    // leaky_relu(0.2)
                ev[cnt] = e; sr[cnt] = s; mx = fmaxf(mx, e); ++cnt;
            }
        float sum = 0.0f;
        for (int i = 0; i < cnt; ++i) { ev[i] = __expf(ev[i] - mx); sum += ev[i]; }
        float inv = 1.0f / sum;
        for (int i = 0; i < cnt; ++i) { alpha[i] = ev[i] * inv; srcrow[i] = sr[i]; }
        nvalid = cnt;
    }
    __syncthreads();
    int cnt = nvalid;
    for (int d = threadIdx.x; d < EMBED; d += 256) {
        float acc = 0.0f;
        for (int i = 0; i < cnt; ++i)
            acc += alpha[i] * H[(size_t)srcrow[i] * (4 * EMBED) + gh * EMBED + d];
        float v = acc + gbias[gh * EMBED + d];
        v = (v > 0.0f) ? v : expm1f(v);            // elu
        OG[(size_t)rn * (4 * EMBED) + gh * EMBED + d] = (__bf16)v;
    }
}

// ---------------- 1x1 head over channels ----------------
__global__ void k_head(const float* __restrict__ OG2, const float* __restrict__ hw,
                       const float* __restrict__ hb, float* __restrict__ out) {
    __shared__ float red[256];
    int row = blockIdx.x;                         // b*196 + p
    float s = 0.0f;
    for (int i = threadIdx.x; i < EMBED; i += 256)
        s += OG2[(size_t)row * EMBED + i] * hw[i];
    float r = block_reduce_sum(s, red);
    if (threadIdx.x == 0) out[row] = r + hb[0];
}

// ---------------- host ----------------
static void launch_gemm(hipStream_t st, const void* A, const float* W, const float* bias,
                        void* C, int N, int K, int act, int outbf) {
    dim3 g(N / 64, MPAD / 64), b(128, 1, 1);
    const __bf16* Ab = (const __bf16*)A;
    if (act == 0 && outbf == 0)      k_gemm<0, 0><<<g, b, 0, st>>>(Ab, W, bias, C, N, K);
    else if (act == 0 && outbf == 1) k_gemm<0, 1><<<g, b, 0, st>>>(Ab, W, bias, C, N, K);
    else if (act == 1 && outbf == 0) k_gemm<1, 0><<<g, b, 0, st>>>(Ab, W, bias, C, N, K);
    else                             k_gemm<1, 1><<<g, b, 0, st>>>(Ab, W, bias, C, N, K);
}

extern "C" void kernel_launch(void* const* d_in, const int* in_sizes, int n_in,
                              void* d_out, int out_size, void* d_ws, size_t ws_size,
                              hipStream_t stream) {
    (void)in_sizes; (void)n_in; (void)out_size; (void)ws_size;

    const float* x       = (const float*)d_in[0];
    const float* patch_w = (const float*)d_in[1];
    const float* patch_b = (const float*)d_in[2];
    const float* clsw    = (const float*)d_in[3];
    const float* pos     = (const float*)d_in[4];
    const float* ln1_g   = (const float*)d_in[5];
    const float* ln1_b   = (const float*)d_in[6];
    const float* qkv_w   = (const float*)d_in[7];
    const float* qkv_b   = (const float*)d_in[8];
    const float* proj_w  = (const float*)d_in[9];
    const float* proj_b  = (const float*)d_in[10];
    const float* ln2_g   = (const float*)d_in[11];
    const float* ln2_b   = (const float*)d_in[12];
    const float* fc1_w   = (const float*)d_in[13];
    const float* fc1_b   = (const float*)d_in[14];
    const float* fc2_w   = (const float*)d_in[15];
    const float* fc2_b   = (const float*)d_in[16];
    const float* norm_g  = (const float*)d_in[17];
    const float* norm_b  = (const float*)d_in[18];
    const float* spat_w  = (const float*)d_in[19];
    const float* spat_b  = (const float*)d_in[20];
    const float* gauss   = (const float*)d_in[21];
    const float* gat_w   = (const float*)d_in[22];
    const float* a_src   = (const float*)d_in[23];
    const float* a_dst   = (const float*)d_in[24];
    const float* gat_b   = (const float*)d_in[25];
    const float* gproj_w = (const float*)d_in[26];
    const float* gproj_b = (const float*)d_in[27];
    const float* head_w  = (const float*)d_in[28];
    const float* head_b  = (const float*)d_in[29];

    // workspace layout (phase-aliased): ~74 MB total
    char* ws = (char*)d_ws;
    size_t off = 0;
    auto take = [&](size_t bytes) -> char* {
        char* p = ws + off;
        off += (bytes + 255) & ~(size_t)255;
        return p;
    };
    float* H    = (float*)take((size_t)MPAD * EMBED * 4);        // residual stream
    float* Y    = (float*)take((size_t)MPAD * EMBED * 4);        // post-norm stream
    char*  buf1 = take((size_t)MPAD * (EMBED * 9) * 2);          // im2col(patch/sp), LN-out bf16
    char*  buf2 = take((size_t)MPAD * MLPDIM * 4);               // qkv f32 | hgat f32
    char*  buf3 = take((size_t)MPAD * EMBED * 4);                // patchout | attn-out bf16 | spout
    char*  buf4 = take((size_t)MPAD * EMBED * 4);                // proj/fc2 out | og2
    char*  buf5 = take((size_t)MPAD * MLPDIM * 2);               // fc1 bf16 | og bf16
    char*  buf6 = take((size_t)MPAD * EMBED * 2);                // nodes bf16
    float* SS   = (float*)take((size_t)MREAL_P * 4 * 4);
    float* SD   = (float*)take((size_t)MREAL_P * 4 * 4);

    auto cdiv = [](int a, int b) { return (a + b - 1) / b; };
    float* outMap = (float*)d_out;               // [0,1568): out ; [1568,1568+6144): cls_out
    float* clsOut = outMap + MREAL_P;

    // ---- patch embedding (conv16x16 s16 as WMMA GEMM) ----
    k_im2col_patch<<<cdiv(MREAL_P * EMBED, 256), 256, 0, stream>>>(x, (__bf16*)buf1);
    launch_gemm(stream, buf1, patch_w, patch_b, buf3, EMBED, EMBED, 0, 0);
    k_build_h<<<cdiv(MREAL_T * EMBED, 256), 256, 0, stream>>>((float*)buf3, clsw, pos, H);

    // ---- 12 transformer blocks ----
    for (int l = 0; l < 12; ++l) {
        k_layernorm<<<MREAL_T, 256, 0, stream>>>(H, ln1_g + (size_t)l * EMBED,
                                                 ln1_b + (size_t)l * EMBED, (__bf16*)buf1);
        launch_gemm(stream, buf1, qkv_w + (size_t)l * 3 * EMBED * EMBED,
                    qkv_b + (size_t)l * 3 * EMBED, buf2, 3 * EMBED, EMBED, 0, 0);
        k_attention<<<BATCH * NHEADS * TOKENS, 256, 0, stream>>>((float*)buf2, (__bf16*)buf3);
        launch_gemm(stream, buf3, proj_w + (size_t)l * EMBED * EMBED,
                    proj_b + (size_t)l * EMBED, buf4, EMBED, EMBED, 0, 0);
        k_resadd<<<cdiv(MREAL_T * EMBED, 256), 256, 0, stream>>>(H, (float*)buf4, MREAL_T * EMBED);

        k_layernorm<<<MREAL_T, 256, 0, stream>>>(H, ln2_g + (size_t)l * EMBED,
                                                 ln2_b + (size_t)l * EMBED, (__bf16*)buf1);
        launch_gemm(stream, buf1, fc1_w + (size_t)l * MLPDIM * EMBED,
                    fc1_b + (size_t)l * MLPDIM, buf5, MLPDIM, EMBED, 1, 1);   // GELU -> bf16
        launch_gemm(stream, buf5, fc2_w + (size_t)l * EMBED * MLPDIM,
                    fc2_b + (size_t)l * EMBED, buf4, EMBED, MLPDIM, 0, 0);
        k_resadd<<<cdiv(MREAL_T * EMBED, 256), 256, 0, stream>>>(H, (float*)buf4, MREAL_T * EMBED);
    }

    // ---- final norms, +pos, norm; cls_out ----
    k_final_norm<<<MREAL_T, 256, 0, stream>>>(H, norm_g, norm_b, pos, Y, clsOut);

    // ---- spatial 3x3 conv as WMMA GEMM (K=6912) + GELU ----
    k_im2col_sp<<<cdiv(MREAL_P * EMBED * 9, 256), 256, 0, stream>>>(Y, (__bf16*)buf1);
    launch_gemm(stream, buf1, spat_w, spat_b, buf3, EMBED, EMBED * 9, 1, 0);
    k_pe_add<<<cdiv(MREAL_P * EMBED, 256), 256, 0, stream>>>((float*)buf3, gauss, (__bf16*)buf6);

    // ---- GAT ----
    launch_gemm(stream, buf6, gat_w, nullptr, buf2, 4 * EMBED, EMBED, 0, 0);     // hgat f32
    k_gat_scores<<<MREAL_P, 256, 0, stream>>>((float*)buf2, a_src, a_dst, SS, SD);
    k_gat_agg<<<MREAL_P * 4, 256, 0, stream>>>((float*)buf2, SS, SD, gat_b, (__bf16*)buf5);
    launch_gemm(stream, buf5, gproj_w, gproj_b, buf4, EMBED, 4 * EMBED, 0, 0);   // og2 f32

    // ---- head ----
    k_head<<<MREAL_P, 256, 0, stream>>>((float*)buf4, head_w, head_b, outMap);
}
